// GNNEncoder_67027259621724
// MI455X (gfx1250) — matrix operand order repaired and checked by reading, not compiled, not run on hardware
//
#include <hip/hip_runtime.h>
#include <hip/hip_bf16.h>

// ---------------- problem constants (from reference) ----------------
#define F_IN 128
#define HDIM 256
#define LDIM 64
#define GNUM 500

typedef __attribute__((ext_vector_type(16))) __bf16 v16bf;
typedef __attribute__((ext_vector_type(8)))  float  v8f;

union BF16x16 {
    v16bf v;
    unsigned int   u[8];
    unsigned short s[16];
};

__device__ __forceinline__ unsigned short f2bf(float f) {
    union { float f; unsigned int u; } c; c.f = f;
    unsigned int u = c.u;
    u += 0x7FFFu + ((u >> 16) & 1u);   // round-to-nearest-even
    return (unsigned short)(u >> 16);
}

// Relaxed, agent-scope fp32 atomic add -> native global_atomic_add_f32 on CDNA5
__device__ __forceinline__ void atomic_add_f32(float* p, float v) {
#if __has_builtin(__hip_atomic_fetch_add)
    __hip_atomic_fetch_add(p, v, __ATOMIC_RELAXED, __HIP_MEMORY_SCOPE_AGENT);
#else
    atomicAdd(p, v);
#endif
}

// ---------------- degree / norm ----------------
__global__ void k_deg_init(float* deg, int n) {
    int i = blockIdx.x * blockDim.x + threadIdx.x;
    if (i < n) deg[i] = 1.0f;                      // self loop
}

__global__ void k_deg_accum(const int* __restrict__ ei, float* deg, int nE) {
    int e = blockIdx.x * blockDim.x + threadIdx.x;
    if (e < nE) atomic_add_f32(&deg[ei[nE + e]], 1.0f); // dst = edge_index[1]
}

__global__ void k_dinv(float* deg, int n) {
    int i = blockIdx.x * blockDim.x + threadIdx.x;
    if (i < n) {
        float d = deg[i];
        deg[i] = d > 0.0f ? rsqrtf(d) : 0.0f;      // in-place -> dinv
    }
}

// ---------------- weight transpose + bf16 convert: W[K][H] -> Wt[H][K] ----------------
__global__ void k_conv_weight(const float* __restrict__ W, unsigned short* __restrict__ Wt,
                              int K, int Hc) {
    int idx = blockIdx.x * blockDim.x + threadIdx.x;
    if (idx >= K * Hc) return;
    int k = idx / Hc;
    int h = idx % Hc;
    Wt[(size_t)h * K + k] = f2bf(W[idx]);
}

// ---------------- WMMA GEMM: C[nrows x 256] = A[nrows x K] (f32, cvt bf16) @ Wt^T ----------------
// Wt is [256][K] bf16 (column-of-W contiguous). One wave computes a 16x64 output strip.
__global__ void k_wmma_gemm(const float* __restrict__ A, const unsigned short* __restrict__ Wt,
                            float* __restrict__ C, int nrows, int K) {
    int wave = (blockIdx.x * blockDim.x + threadIdx.x) >> 5;
    int lane = threadIdx.x & 31;
    int m_tile  = wave >> 2;                 // 16-row tile index
    int n_group = wave & 3;                  // 64-col group
    int tiles_m = nrows >> 4;
    if (m_tile >= tiles_m) return;

    int row  = m_tile * 16 + (lane & 15);
    int kbA  = 16 * (lane >> 4);             // A: contiguous k = 16h + s
    int kbB  = 16 * (lane >> 4);             // B: contiguous k = 16h + s, same as A

    v8f acc0 = {}, acc1 = {}, acc2 = {}, acc3 = {};

    const float* arow = A + (size_t)row * K;

    for (int k0 = 0; k0 < K; k0 += 32) {
        // ---- A fragment: elems 0-7 -> K=k0+kbA.., elems 8-15 -> K=k0+kbA+8.. ----
        const float4* ap0 = reinterpret_cast<const float4*>(arow + k0 + kbA);
        const float4* ap1 = reinterpret_cast<const float4*>(arow + k0 + kbA + 8);
        float4 f0 = ap0[0], f1 = ap0[1];
        float4 f2 = ap1[0], f3 = ap1[1];
        BF16x16 af;
        af.s[0]  = f2bf(f0.x); af.s[1]  = f2bf(f0.y); af.s[2]  = f2bf(f0.z); af.s[3]  = f2bf(f0.w);
        af.s[4]  = f2bf(f1.x); af.s[5]  = f2bf(f1.y); af.s[6]  = f2bf(f1.z); af.s[7]  = f2bf(f1.w);
        af.s[8]  = f2bf(f2.x); af.s[9]  = f2bf(f2.y); af.s[10] = f2bf(f2.z); af.s[11] = f2bf(f2.w);
        af.s[12] = f2bf(f3.x); af.s[13] = f2bf(f3.y); af.s[14] = f2bf(f3.z); af.s[15] = f2bf(f3.w);

        // ---- B fragments: 16 contiguous bf16 per lane at Wt[col][k0+kbB..] ----
        #pragma unroll
        for (int t = 0; t < 4; ++t) {
            int col = n_group * 64 + t * 16 + (lane & 15);
            const uint4* bp = reinterpret_cast<const uint4*>(Wt + (size_t)col * K + k0 + kbB);
            uint4 b0 = bp[0], b1 = bp[1];
            BF16x16 bf;
            bf.u[0] = b0.x; bf.u[1] = b0.y; bf.u[2] = b0.z; bf.u[3] = b0.w;
            bf.u[4] = b1.x; bf.u[5] = b1.y; bf.u[6] = b1.z; bf.u[7] = b1.w;
            v8f* accp = (t == 0) ? &acc0 : (t == 1) ? &acc1 : (t == 2) ? &acc2 : &acc3;
            *accp = __builtin_amdgcn_wmma_f32_16x16x32_bf16(
                        false, af.v, false, bf.v, (short)0, *accp, false, false);
        }
    }

    // ---- store: lane 0-15 -> rows 0-7 (elem v = row v), lane 16-31 -> rows 8-15 ----
    int rbase = m_tile * 16 + 8 * (lane >> 4);
    #pragma unroll
    for (int t = 0; t < 4; ++t) {
        int col = n_group * 64 + t * 16 + (lane & 15);
        v8f acc = (t == 0) ? acc0 : (t == 1) ? acc1 : (t == 2) ? acc2 : acc3;
        float* cp = C + (size_t)rbase * HDIM + col;
        #pragma unroll
        for (int v = 0; v < 8; ++v) cp[(size_t)v * HDIM] = acc[v];
    }
}

// ---------------- agg[i][:] = dinv[i]^2 * T[i][:]  (self-loop term) ----------------
__global__ void k_selfloop_init(const float* __restrict__ dinv, const float* __restrict__ T,
                                float* __restrict__ Agg, int total) {
    int i = blockIdx.x * blockDim.x + threadIdx.x;
    if (i >= total) return;
    int node = i >> 8;                       // /HDIM
    float w = dinv[node];
    Agg[i] = (w * w) * T[i];
}

// ---------------- edge scatter: one wave per edge, 8 floats per lane ----------------
__global__ void k_scatter_edges(const int* __restrict__ ei, const float* __restrict__ dinv,
                                const float* __restrict__ T, float* __restrict__ Agg, int nE) {
    int gid  = blockIdx.x * blockDim.x + threadIdx.x;
    int edge = gid >> 5;
    int lane = gid & 31;
    if (edge >= nE) return;
    int s = ei[edge];
    int d = ei[nE + edge];
    float w = dinv[s] * dinv[d];
    const float4* sp = reinterpret_cast<const float4*>(T + (size_t)s * HDIM) + lane * 2;
    float*        dp = Agg + (size_t)d * HDIM + lane * 8;
    float4 v0 = sp[0], v1 = sp[1];
    atomic_add_f32(dp + 0, w * v0.x); atomic_add_f32(dp + 1, w * v0.y);
    atomic_add_f32(dp + 2, w * v0.z); atomic_add_f32(dp + 3, w * v0.w);
    atomic_add_f32(dp + 4, w * v1.x); atomic_add_f32(dp + 5, w * v1.y);
    atomic_add_f32(dp + 6, w * v1.z); atomic_add_f32(dp + 7, w * v1.w);
}

// ---------------- h = relu(agg + b) ----------------
__global__ void k_finalize_relu(const float* __restrict__ Agg, const float* __restrict__ b,
                                float* __restrict__ Hout, int total) {
    int i = blockIdx.x * blockDim.x + threadIdx.x;
    if (i >= total) return;
    int col = i & (HDIM - 1);
    Hout[i] = fmaxf(Agg[i] + b[col], 0.0f);
}

// ---------------- zero pooling buffers ----------------
__global__ void k_zero(float* p, int n) {
    int i = blockIdx.x * blockDim.x + threadIdx.x;
    if (i < n) p[i] = 0.0f;
}

__global__ void k_count_nodes(const int* __restrict__ batch, float* cnt, int n) {
    int i = blockIdx.x * blockDim.x + threadIdx.x;
    if (i < n) atomic_add_f32(&cnt[batch[i]], 1.0f);
}

// ---------------- fused relu + pooled sums ----------------
__global__ void k_finalize_pool(const float* __restrict__ Agg, const float* __restrict__ b,
                                const int* __restrict__ batch, float* __restrict__ sums, int total) {
    int i = blockIdx.x * blockDim.x + threadIdx.x;
    if (i >= total) return;
    int col  = i & (HDIM - 1);
    int node = i >> 8;
    float v = fmaxf(Agg[i] + b[col], 0.0f);
    atomic_add_f32(&sums[(size_t)batch[node] * HDIM + col], v);
}

// ---------------- per-graph head: hg = sums/cnt; mu/logvar = hg @ W + b ----------------
__global__ void k_pool_head(const float* __restrict__ sums, const float* __restrict__ cnt,
                            const float* __restrict__ Wmu, const float* __restrict__ bmu,
                            const float* __restrict__ Wlv, const float* __restrict__ blv,
                            float* __restrict__ out) {
    __shared__ float hg[HDIM];
    int g = blockIdx.x;
    float c = fmaxf(cnt[g], 1.0f);
    for (int i = threadIdx.x; i < HDIM; i += blockDim.x)
        hg[i] = sums[(size_t)g * HDIM + i] / c;
    __syncthreads();
    int t = threadIdx.x;
    if (t < LDIM) {
        float acc = bmu[t];
        #pragma unroll 4
        for (int k = 0; k < HDIM; ++k) acc += hg[k] * Wmu[k * LDIM + t];
        out[(size_t)g * LDIM + t] = acc;                               // mu
    } else if (t < 2 * LDIM) {
        int l = t - LDIM;
        float acc = blv[l];
        #pragma unroll 4
        for (int k = 0; k < HDIM; ++k) acc += hg[k] * Wlv[k * LDIM + l];
        out[(size_t)GNUM * LDIM + (size_t)g * LDIM + l] = acc;         // logvar
    }
}

// ---------------- launch ----------------
extern "C" void kernel_launch(void* const* d_in, const int* in_sizes, int n_in,
                              void* d_out, int out_size, void* d_ws, size_t ws_size,
                              hipStream_t stream) {
    const float* x    = (const float*)d_in[0];
    const int*   ei   = (const int*)  d_in[1];
    const int*   batch= (const int*)  d_in[2];
    const float* W1   = (const float*)d_in[4];
    const float* b1   = (const float*)d_in[5];
    const float* W2   = (const float*)d_in[6];
    const float* b2   = (const float*)d_in[7];
    const float* Wmu  = (const float*)d_in[8];
    const float* bmu  = (const float*)d_in[9];
    const float* Wlv  = (const float*)d_in[10];
    const float* blv  = (const float*)d_in[11];
    float* out = (float*)d_out;

    const int N  = in_sizes[0] / F_IN;
    const int E  = in_sizes[1] / 2;
    const int NH = N * HDIM;

    // workspace carving (256B aligned)
    char* w = (char*)d_ws;
    auto carve = [&](size_t bytes) {
        void* p = (void*)w;
        w += (bytes + 255) & ~(size_t)255;
        return p;
    };
    float*          dinv  = (float*)         carve((size_t)N * 4);
    unsigned short* Wt1   = (unsigned short*)carve((size_t)HDIM * F_IN * 2);
    unsigned short* Wt2   = (unsigned short*)carve((size_t)HDIM * HDIM * 2);
    float*          bufT  = (float*)         carve((size_t)NH * 4);
    float*          bufAgg= (float*)         carve((size_t)NH * 4);
    float*          sums  = (float*)         carve((size_t)GNUM * HDIM * 4);
    float*          cnt   = (float*)         carve((size_t)GNUM * 4);

    const int T = 256;
    auto cdiv = [](int a, int b) { return (a + b - 1) / b; };

    // 1) normalization coefficients
    k_deg_init  <<<cdiv(N, T), T, 0, stream>>>(dinv, N);
    k_deg_accum <<<cdiv(E, T), T, 0, stream>>>(ei, dinv, E);
    k_dinv      <<<cdiv(N, T), T, 0, stream>>>(dinv, N);

    // 2) weight prep (transpose + bf16)
    k_conv_weight<<<cdiv(F_IN * HDIM, T), T, 0, stream>>>(W1, Wt1, F_IN, HDIM);
    k_conv_weight<<<cdiv(HDIM * HDIM, T), T, 0, stream>>>(W2, Wt2, HDIM, HDIM);

    const int gemm_waves  = (N / 16) * 4;
    const int gemm_blocks = cdiv(gemm_waves * 32, T);

    // 3) layer 1: t1 = x @ W1 ; agg = self + scatter ; h1 = relu(agg + b1)
    k_wmma_gemm    <<<gemm_blocks, T, 0, stream>>>(x, Wt1, bufT, N, F_IN);
    k_selfloop_init<<<cdiv(NH, T), T, 0, stream>>>(dinv, bufT, bufAgg, NH);
    k_scatter_edges<<<cdiv(E * 32, T), T, 0, stream>>>(ei, dinv, bufT, bufAgg, E);
    k_finalize_relu<<<cdiv(NH, T), T, 0, stream>>>(bufAgg, b1, bufT, NH);

    // 4) layer 2: t2 = h1 @ W2 ; agg ; pooled relu
    k_wmma_gemm    <<<gemm_blocks, T, 0, stream>>>(bufT, Wt2, bufAgg, N, HDIM);
    k_selfloop_init<<<cdiv(NH, T), T, 0, stream>>>(dinv, bufAgg, bufT, NH);
    k_scatter_edges<<<cdiv(E * 32, T), T, 0, stream>>>(ei, dinv, bufAgg, bufT, E);

    // 5) pooling + heads
    k_zero         <<<cdiv(GNUM * HDIM, T), T, 0, stream>>>(sums, GNUM * HDIM);
    k_zero         <<<cdiv(GNUM, T), T, 0, stream>>>(cnt, GNUM);
    k_count_nodes  <<<cdiv(N, T), T, 0, stream>>>(batch, cnt, N);
    k_finalize_pool<<<cdiv(NH, T), T, 0, stream>>>(bufT, b2, batch, sums, NH);
    k_pool_head    <<<GNUM, 128, 0, stream>>>(sums, cnt, Wmu, bmu, Wlv, blv, out);
}
